// STAttentionBlock_88201448391293
// MI455X (gfx1250) — matrix operand order repaired and checked by reading, not compiled, not run on hardware
//
#include <hip/hip_runtime.h>
#include <hip/hip_bf16.h>

// ST-Attention block for MI455X (gfx1250), bf16 WMMA pipeline.
// All heavy contractions lower to v_wmma_f32_16x16x32_bf16 (wave32, fp32 acc).
// Memory-bound problem: bf16 working set (~160MB) kept L2-resident (192MB L2).

typedef __bf16 bf16;
typedef __attribute__((ext_vector_type(16))) __bf16 v16bf;
typedef __attribute__((ext_vector_type(4)))  __bf16 v4bf;
typedef __attribute__((ext_vector_type(8)))  float  f32x8;

#define WMMA_BF16(acc, a, b) \
  (acc) = __builtin_amdgcn_wmma_f32_16x16x32_bf16(false, (a), false, (b), (short)0, (acc), false, false)

#define N_   256
#define C_   64
#define T_   64
#define V_   25
#define S_   3
#define IC_  16
#define TV_  1600
#define CTV_ 102400   // C_*TV_ per sample
#define NWAVES 8

// ---- workspace layout (bytes) ----
#define OFF_XB     ((size_t)0)            // bf16 [N][64][1600]  x, later overwritten by y2 (in place)
#define OFF_Y      ((size_t)52428800)     // bf16 [N][64][1600]  y / z1
#define OFF_STAGE  ((size_t)104857600)    // bf16 [N][64][1600]  pre-BN conv outputs
#define OFF_ATT    ((size_t)157286400)    // f32  [N][3][64*64]  attention (spatial uses 25x25 prefix)
#define OFF_X1S    ((size_t)169869312)    // bf16 [3][64][1600]
#define OFF_X1T    ((size_t)170483712)    // bf16 [3][64][1600]
#define OFF_QKPES  ((size_t)171098112)    // f32  [96][1600]  Win_s@pe_s + bin_s
#define OFF_QKPET  ((size_t)171712512)    // f32  [96][1600]
#define OFF_WINS   ((size_t)172326912)    // bf16 [96][64]
#define OFF_WINT   ((size_t)172339200)
#define OFF_WOUTS  ((size_t)172351488)    // bf16 [64][192]
#define OFF_WOUTT  ((size_t)172376064)
#define OFF_WFFS   ((size_t)172400640)    // bf16 [64][64]
#define OFF_WFFT   ((size_t)172408832)
#define OFF_STATS  ((size_t)172417024)    // f32  4 stages x 128 (sum,sumsq per channel)

// A-fragment K offset for 16-bit 16x32 A tiles (ISA 7.12.2):
// lanes 0-15: K = {0..7,16..23}; lanes 16-31: +8
__device__ __forceinline__ int ak_off(int lane, int h) {
  return ((h < 8) ? h : (h + 8)) + ((lane & 16) ? 8 : 0);
}
// B-fragment: same K order as A (lanes 0-15: K = {0..7,16..23}; lanes 16-31: +8), column = lane&15
__device__ __forceinline__ int bk_off(int lane, int h) {
  return ((h < 8) ? h : (h + 8)) + ((lane & 16) ? 8 : 0);
}

// ---------------- utility kernels ----------------
__global__ __launch_bounds__(256) void k_zero(float* p, int n) {
  int i = blockIdx.x * 256 + threadIdx.x;
  if (i < n) p[i] = 0.f;
}

// n must be a multiple of 4 (all our tensors are). float4 in, 4x bf16 out.
__global__ __launch_bounds__(256) void k_castf2b(const float* __restrict__ src,
                                                 bf16* __restrict__ dst, int n4) {
  const float4* s4 = (const float4*)src;
  v4bf* d4 = (v4bf*)dst;
  for (int i = blockIdx.x * 256 + threadIdx.x; i < n4; i += gridDim.x * 256) {
    float4 s = s4[i];
    v4bf d = {(bf16)s.x, (bf16)s.y, (bf16)s.z, (bf16)s.w};
    d4[i] = d;
  }
}

// qkpe[o][tv] = bin[o] + sum_c W[o][c] * pe[c][tv]   (96 x 1600, tiny, n-independent)
__global__ __launch_bounds__(256) void k_qkpe(const float* __restrict__ W,
                                              const float* __restrict__ pe,
                                              const float* __restrict__ bias,
                                              float* __restrict__ out) {
  int i = blockIdx.x * 256 + threadIdx.x;   // 96*1600 threads
  int o = i / TV_, tv = i % TV_;
  float acc = bias[o];
  for (int c = 0; c < C_; ++c) acc += W[o * C_ + c] * pe[c * TV_ + tv];
  out[i] = acc;
}

// ---------------- attention kernel (spatial & temporal) ----------------
// block = (s, n). LDS: full sample (64x1600 bf16) + Q,K (16x1600 bf16 each) + Win rows.
// Phase 1: Q/K = Win * x   (pe handled via qkpe table; conv is linear)
// Phase 2: att = tanh(Q.K / (IC*L))*alpha + tanh(x1 . x)*beta + att0
template <bool TEMP>
__global__ __launch_bounds__(256)
void k_att(const bf16* __restrict__ xin, const bf16* __restrict__ Wb,
           const float* __restrict__ qkpe, const bf16* __restrict__ x1b,
           const float* __restrict__ att0, const float* __restrict__ alpha,
           const float* __restrict__ beta, float* __restrict__ attout)
{
  extern __shared__ char smem[];
  bf16* xl = (bf16*)smem;        // 64*1600
  bf16* Ql = xl + CTV_;          // 16*1600
  bf16* Kl = Ql + IC_ * TV_;     // 16*1600
  bf16* aW = Kl + IC_ * TV_;     // 32*64
  const int tid = threadIdx.x, lane = tid & 31, wave = tid >> 5;
  const int s = blockIdx.x, n = blockIdx.y;

  // b128 cooperative fill of the sample (204800 B = 12800 x uint4)
  {
    const uint4* src4 = (const uint4*)(xin + (size_t)n * CTV_);
    uint4* dst4 = (uint4*)xl;
    for (int i = tid; i < CTV_ / 8; i += 256) dst4[i] = src4[i];
  }
  for (int i = tid; i < 32 * 64; i += 256) {
    int m = i >> 6, c = i & 63;
    int row = (m < 16) ? (s * IC_ + m) : (S_ * IC_ + s * IC_ + (m - 16));
    aW[i] = Wb[row * 64 + c];
  }
  __syncthreads();

  // Phase 1: 2 m-tiles (Q rows / K rows) x 100 n-tiles, K-dim 64 (2 steps)
  for (int tile = wave; tile < 200; tile += NWAVES) {
    int mt = tile & 1, nti = tile >> 1;
    int col = nti * 16 + (lane & 15);
    f32x8 acc = {0.f,0.f,0.f,0.f,0.f,0.f,0.f,0.f};
    for (int kc = 0; kc < 2; ++kc) {
      v16bf a, b;
#pragma unroll
      for (int h = 0; h < 16; ++h) {
        a[h] = aW[(mt * 16 + (lane & 15)) * 64 + kc * 32 + ak_off(lane, h)];
        b[h] = xl[(kc * 32 + bk_off(lane, h)) * TV_ + col];
      }
      WMMA_BF16(acc, a, b);
    }
    bf16* dst = mt ? Kl : Ql;
    const int rowbase = (mt ? S_ * IC_ : 0) + s * IC_;
#pragma unroll
    for (int r = 0; r < 8; ++r) {
      int m8 = r + ((lane & 16) ? 8 : 0);
      dst[m8 * TV_ + col] = (bf16)(acc[r] + qkpe[(rowbase + m8) * TV_ + col]);
    }
  }
  __syncthreads();

  // Phase 2: attention tiles (spatial 2x2 over 25x25 padded, temporal 4x4 over 64x64)
  const int NT = TEMP ? 4 : 2;
  const float a_s = alpha[s], b_s = beta[s];
  const float scale = TEMP ? (1.0f / 400.0f) : (1.0f / 1024.0f);
  const bf16* x1s = x1b + s * CTV_;
  float* slot = attout + ((size_t)n * S_ + s) * 4096;
  const bf16 Z = (bf16)0.0f;

  for (int tile = wave; tile < NT * NT; tile += NWAVES) {
    int mt = tile / NT, nt = tile % NT;
    int mrow = mt * 16 + (lane & 15);
    int ncol = nt * 16 + (lane & 15);

    // att1 = Q^T K  (spatial: k=(c,t) 1024; temporal: k=(c,v) 400 padded 416)
    f32x8 acc1 = {0.f,0.f,0.f,0.f,0.f,0.f,0.f,0.f};
    const int K1S = TEMP ? 13 : 32;
    for (int kc = 0; kc < K1S; ++kc) {
      v16bf a, b;
#pragma unroll
      for (int h = 0; h < 16; ++h) {
        int ka = kc * 32 + ak_off(lane, h);
        int kb = kc * 32 + bk_off(lane, h);
        if (TEMP) {
          a[h] = (ka < 400) ? Ql[(ka / 25) * TV_ + mrow * 25 + (ka % 25)] : Z;
          b[h] = (kb < 400) ? Kl[(kb / 25) * TV_ + ncol * 25 + (kb % 25)] : Z;
        } else {
          a[h] = (mrow < 25) ? Ql[(ka >> 6) * TV_ + (ka & 63) * 25 + mrow] : Z;
          b[h] = (ncol < 25) ? Kl[(kb >> 6) * TV_ + (kb & 63) * 25 + ncol] : Z;
        }
      }
      WMMA_BF16(acc1, a, b);
    }
    // att2 = x1^T x  (spatial k=(c,t) 4096; temporal k=(c,v) 1600). A from global (L2-hot).
    f32x8 acc2 = {0.f,0.f,0.f,0.f,0.f,0.f,0.f,0.f};
    const int K2S = TEMP ? 50 : 128;
    for (int kc = 0; kc < K2S; ++kc) {
      v16bf a, b;
#pragma unroll
      for (int h = 0; h < 16; ++h) {
        int ka = kc * 32 + ak_off(lane, h);
        int kb = kc * 32 + bk_off(lane, h);
        if (TEMP) {
          a[h] = x1s[(ka / 25) * TV_ + mrow * 25 + (ka % 25)];
          b[h] = xl[(kb / 25) * TV_ + ncol * 25 + (kb % 25)];
        } else {
          a[h] = (mrow < 25) ? x1s[(ka >> 6) * TV_ + (ka & 63) * 25 + mrow] : Z;
          b[h] = (ncol < 25) ? xl[(kb >> 6) * TV_ + (kb & 63) * 25 + ncol] : Z;
        }
      }
      WMMA_BF16(acc2, a, b);
    }
#pragma unroll
    for (int r = 0; r < 8; ++r) {
      int mr = mt * 16 + r + ((lane & 16) ? 8 : 0);
      float v = tanhf(acc1[r] * scale) * a_s + tanhf(acc2[r]) * b_s;
      if (TEMP) {
        slot[mr * 64 + ncol] = v + att0[s * 4096 + mr * 64 + ncol];
      } else if (mr < 25 && ncol < 25) {
        slot[mr * 25 + ncol] = v + att0[s * 625 + mr * 25 + ncol];
      }
    }
  }
}

// ------------- spatial attention-apply + Wout_s conv + BN stats -------------
// block = (tgroup of 8 t's, n). y[s,(c,t),v] = x[(c,t),u] att[s,u,v]; then 64x192 conv.
__global__ __launch_bounds__(256)
void k_y_s(const bf16* __restrict__ xb, const float* __restrict__ attg,
           const bf16* __restrict__ Wb, const float* __restrict__ bias,
           bf16* __restrict__ stage, float* __restrict__ stats)
{
  extern __shared__ char smem[];
  bf16*  xs    = (bf16*)smem;              // [512][32] (u padded to 32)       32768 B
  bf16*  att3b = (bf16*)(smem + 32768);    // [3][32][32] bf16, zero-padded     6144 B
  bf16*  ysl   = (bf16*)(smem + 38912);    // [3][512][25]                     76800 B
  bf16*  Wl    = (bf16*)(smem + 115712);   // [64][192]                        24576 B
  float* stl   = (float*)(smem + 140288);  // [128]                              512 B
  const int tid = threadIdx.x, lane = tid & 31, wave = tid >> 5;
  const int tg = blockIdx.x, n = blockIdx.y, t0 = tg * 8;
  const bf16* xg = xb + (size_t)n * CTV_;

  for (int i = tid; i < 512 * 32; i += 256) {
    int m = i >> 5, u = i & 31, c = m >> 3, tl = m & 7;
    xs[i] = (u < 25) ? xg[c * TV_ + (t0 + tl) * 25 + u] : (bf16)0.0f;
  }
  for (int i = tid; i < 3 * 1024; i += 256) {
    int s = i >> 10, rest = i & 1023, u = rest >> 5, v = rest & 31;
    att3b[i] = (u < 25 && v < 25) ? (bf16)attg[((size_t)n * S_ + s) * 4096 + u * 25 + v]
                                  : (bf16)0.0f;
  }
  {
    const uint4* w4 = (const uint4*)Wb;
    uint4* d4 = (uint4*)Wl;
    for (int i = tid; i < (64 * 192) / 8; i += 256) d4[i] = w4[i];
  }
  if (tid < 128) stl[tid] = 0.f;
  __syncthreads();

  // ys: 3 s x 32 m-tiles x 2 v-tiles, single K step (u padded to 32)
  for (int tile = wave; tile < 192; tile += NWAVES) {
    int s = tile / 64, rem = tile % 64, mtl = rem >> 1, nt = rem & 1;
    int m = mtl * 16 + (lane & 15), vcol = nt * 16 + (lane & 15);
    f32x8 acc = {0.f,0.f,0.f,0.f,0.f,0.f,0.f,0.f};
    v16bf a, b;
#pragma unroll
    for (int h = 0; h < 16; ++h) {
      a[h] = xs[m * 32 + ak_off(lane, h)];
      b[h] = att3b[s * 1024 + bk_off(lane, h) * 32 + vcol];
    }
    WMMA_BF16(acc, a, b);
#pragma unroll
    for (int r = 0; r < 8; ++r) {
      int mr = mtl * 16 + r + ((lane & 16) ? 8 : 0);
      if (vcol < 25) ysl[s * 12800 + mr * 25 + vcol] = (bf16)acc[r];
    }
  }
  __syncthreads();

  // conv: [64][192] x [192][200]
  for (int tile = wave; tile < 52; tile += NWAVES) {
    int mtl = tile / 13, nt = tile % 13;
    int o = mtl * 16 + (lane & 15), col = nt * 16 + (lane & 15);
    f32x8 acc = {0.f,0.f,0.f,0.f,0.f,0.f,0.f,0.f};
    for (int kc = 0; kc < 6; ++kc) {
      v16bf a, b;
#pragma unroll
      for (int h = 0; h < 16; ++h) {
        a[h] = Wl[o * 192 + kc * 32 + ak_off(lane, h)];
        int kb = kc * 32 + bk_off(lane, h);
        b[h] = (col < 200) ? ysl[(kb >> 6) * 12800 + (kb & 63) * 200 + col] : (bf16)0.0f;
      }
      WMMA_BF16(acc, a, b);
    }
#pragma unroll
    for (int r = 0; r < 8; ++r) {
      int orow = mtl * 16 + r + ((lane & 16) ? 8 : 0);
      if (col < 200) {
        float val = acc[r] + bias[orow];
        stage[(size_t)n * CTV_ + orow * TV_ + t0 * 25 + col] = (bf16)val;
        atomicAdd(&stl[orow * 2], val);
        atomicAdd(&stl[orow * 2 + 1], val * val);
      }
    }
  }
  __syncthreads();
  if (tid < 64) {
    atomicAdd(&stats[tid], stl[tid * 2]);
    atomicAdd(&stats[64 + tid], stl[tid * 2 + 1]);
  }
}

// ------------- temporal attention-apply + Wout_t conv + BN stats -------------
// block = (qgroup of 16 q's, n). z[s,(c,v),q] = y[(c,v),t] attt[s,t,q]; then 64x192 conv.
__global__ __launch_bounds__(256)
void k_y_t(const bf16* __restrict__ yb, const float* __restrict__ attg,
           const bf16* __restrict__ Wb, const float* __restrict__ bias,
           bf16* __restrict__ stage, float* __restrict__ stats)
{
  extern __shared__ char smem[];
  bf16*  zsl   = (bf16*)smem;               // [3][1600][16]   153600 B
  bf16*  attlb = (bf16*)(smem + 153600);    // [3][64][16] bf16  6144 B
  bf16*  Wl    = (bf16*)(smem + 159744);    // [64][192]        24576 B
  float* stl   = (float*)(smem + 184320);   // [128]              512 B
  const int tid = threadIdx.x, lane = tid & 31, wave = tid >> 5;
  const int qg = blockIdx.x, n = blockIdx.y, q0 = qg * 16;
  const bf16* yn = yb + (size_t)n * CTV_;

  for (int i = tid; i < 3 * 1024; i += 256) {
    int s = i >> 10, rest = i & 1023, t = rest >> 4, ql = rest & 15;
    attlb[i] = (bf16)attg[((size_t)n * S_ + s) * 4096 + t * 64 + q0 + ql];
  }
  {
    const uint4* w4 = (const uint4*)Wb;
    uint4* d4 = (uint4*)Wl;
    for (int i = tid; i < (64 * 192) / 8; i += 256) d4[i] = w4[i];
  }
  if (tid < 128) stl[tid] = 0.f;
  __syncthreads();

  // z: 3 s x 100 m-tiles (m=(c,v)), 1 n-tile (16 q's), K-dim 64 (2 steps). A from global (L2/WGP$-hot).
  for (int tile = wave; tile < 300; tile += NWAVES) {
    int s = tile / 100, mtl = tile % 100;
    int m = mtl * 16 + (lane & 15), c = m / 25, v = m % 25, ql = lane & 15;
    f32x8 acc = {0.f,0.f,0.f,0.f,0.f,0.f,0.f,0.f};
    for (int kc = 0; kc < 2; ++kc) {
      v16bf a, b;
#pragma unroll
      for (int h = 0; h < 16; ++h) {
        a[h] = yn[c * TV_ + (kc * 32 + ak_off(lane, h)) * 25 + v];
        b[h] = attlb[s * 1024 + (kc * 32 + bk_off(lane, h)) * 16 + ql];
      }
      WMMA_BF16(acc, a, b);
    }
#pragma unroll
    for (int r = 0; r < 8; ++r) {
      int mr = mtl * 16 + r + ((lane & 16) ? 8 : 0);
      zsl[s * 25600 + mr * 16 + ql] = (bf16)acc[r];
    }
  }
  __syncthreads();

  // conv: [64][192] x [192][400], col=(ql,v)
  for (int tile = wave; tile < 100; tile += NWAVES) {
    int mtl = tile / 25, nt = tile % 25;
    int o = mtl * 16 + (lane & 15), col = nt * 16 + (lane & 15);
    int qlc = col / 25, vc = col % 25;
    f32x8 acc = {0.f,0.f,0.f,0.f,0.f,0.f,0.f,0.f};
    for (int kc = 0; kc < 6; ++kc) {
      v16bf a, b;
#pragma unroll
      for (int h = 0; h < 16; ++h) {
        a[h] = Wl[o * 192 + kc * 32 + ak_off(lane, h)];
        int kb = kc * 32 + bk_off(lane, h);
        b[h] = zsl[(kb / 64) * 25600 + ((kb & 63) * 25 + vc) * 16 + qlc];
      }
      WMMA_BF16(acc, a, b);
    }
#pragma unroll
    for (int r = 0; r < 8; ++r) {
      int orow = mtl * 16 + r + ((lane & 16) ? 8 : 0);
      float val = acc[r] + bias[orow];
      stage[(size_t)n * CTV_ + orow * TV_ + (q0 + qlc) * 25 + vc] = (bf16)val;
      atomicAdd(&stl[orow * 2], val);
      atomicAdd(&stl[orow * 2 + 1], val * val);
    }
  }
  __syncthreads();
  if (tid < 64) {
    atomicAdd(&stats[tid], stl[tid * 2]);
    atomicAdd(&stats[64 + tid], stl[tid * 2 + 1]);
  }
}

// ------------- generic 64x64 conv (Wff) + BN stats -------------
__global__ __launch_bounds__(256)
void k_ff(const bf16* __restrict__ in, const bf16* __restrict__ Wb,
          const float* __restrict__ bias, bf16* __restrict__ stage, float* __restrict__ stats)
{
  extern __shared__ char smem[];
  bf16*  Bc  = (bf16*)smem;             // [64][400]
  bf16*  Wl  = (bf16*)(smem + 51200);   // [64][64]
  float* stl = (float*)(smem + 59392);  // [128]
  const int tid = threadIdx.x, lane = tid & 31, wave = tid >> 5;
  const int cc = blockIdx.x, n = blockIdx.y;
  {
    // per-row b128 fill: 64 rows x 50 uint4 (800 B) each
    uint4* d4 = (uint4*)Bc;
    for (int i = tid; i < 64 * 50; i += 256) {
      int c = i / 50, j = i % 50;
      const uint4* srcrow = (const uint4*)(in + (size_t)n * CTV_ + c * TV_ + cc * 400);
      d4[c * 50 + j] = srcrow[j];
    }
    const uint4* w4 = (const uint4*)Wb;
    uint4* wd = (uint4*)Wl;
    for (int i = tid; i < (64 * 64) / 8; i += 256) wd[i] = w4[i];
  }
  if (tid < 128) stl[tid] = 0.f;
  __syncthreads();

  for (int tile = wave; tile < 100; tile += NWAVES) {
    int mtl = tile / 25, nt = tile % 25;
    int o = mtl * 16 + (lane & 15), col = nt * 16 + (lane & 15);
    f32x8 acc = {0.f,0.f,0.f,0.f,0.f,0.f,0.f,0.f};
    for (int kc = 0; kc < 2; ++kc) {
      v16bf a, b;
#pragma unroll
      for (int h = 0; h < 16; ++h) {
        a[h] = Wl[o * 64 + kc * 32 + ak_off(lane, h)];
        b[h] = Bc[(kc * 32 + bk_off(lane, h)) * 400 + col];
      }
      WMMA_BF16(acc, a, b);
    }
#pragma unroll
    for (int r = 0; r < 8; ++r) {
      int orow = mtl * 16 + r + ((lane & 16) ? 8 : 0);
      float val = acc[r] + bias[orow];
      stage[(size_t)n * CTV_ + orow * TV_ + cc * 400 + col] = (bf16)val;
      atomicAdd(&stl[orow * 2], val);
      atomicAdd(&stl[orow * 2 + 1], val * val);
    }
  }
  __syncthreads();
  if (tid < 64) {
    atomicAdd(&stats[tid], stl[tid * 2]);
    atomicAdd(&stats[64 + tid], stl[tid * 2 + 1]);
  }
}

// ------------- BN apply + residual + LeakyReLU (4 elems/thread, b64/b128 traffic) -------------
template <bool FINAL>
__global__ __launch_bounds__(256)
void k_bn_act(const bf16* __restrict__ stage, const bf16* __restrict__ res,
              const float* __restrict__ stats, const float* __restrict__ w,
              const float* __restrict__ b, bf16* __restrict__ outb, float* __restrict__ outf)
{
  const float invM = 1.0f / 409600.0f;  // N*T*V
  const v4bf* sg4 = (const v4bf*)stage;
  const v4bf* rs4 = (const v4bf*)res;
  for (int i4 = blockIdx.x * 256 + threadIdx.x; i4 < (N_ * CTV_) / 4; i4 += gridDim.x * 256) {
    int o = ((i4 * 4) / TV_) & 63;    // TV_ % 4 == 0 -> channel constant over the 4-group
    float m1 = stats[o] * invM;
    float m2 = stats[64 + o] * invM;
    float rs = rsqrtf(m2 - m1 * m1 + 1e-5f);
    float sw = rs * w[o], sb = b[o] - m1 * rs * w[o];
    v4bf sg = sg4[i4], rr = rs4[i4];
    float t0 = (float)rr[0] + (float)sg[0] * sw + sb;
    float t1 = (float)rr[1] + (float)sg[1] * sw + sb;
    float t2 = (float)rr[2] + (float)sg[2] * sw + sb;
    float t3 = (float)rr[3] + (float)sg[3] * sw + sb;
    t0 = (t0 >= 0.f) ? t0 : 0.1f * t0;
    t1 = (t1 >= 0.f) ? t1 : 0.1f * t1;
    t2 = (t2 >= 0.f) ? t2 : 0.1f * t2;
    t3 = (t3 >= 0.f) ? t3 : 0.1f * t3;
    if (FINAL) {
      float4 o4 = {t0, t1, t2, t3};
      ((float4*)outf)[i4] = o4;
    } else {
      v4bf o4 = {(bf16)t0, (bf16)t1, (bf16)t2, (bf16)t3};
      ((v4bf*)outb)[i4] = o4;
    }
  }
}

// ---------------- host launch ----------------
extern "C" void kernel_launch(void* const* d_in, const int* in_sizes, int n_in,
                              void* d_out, int out_size, void* d_ws, size_t ws_size,
                              hipStream_t stream) {
  const float* x      = (const float*)d_in[0];
  const float* pe_s   = (const float*)d_in[1];
  const float* pe_t   = (const float*)d_in[2];
  const float* Win_s  = (const float*)d_in[3];
  const float* bin_s  = (const float*)d_in[4];
  const float* alphas = (const float*)d_in[5];
  const float* x1_nets= (const float*)d_in[6];
  const float* betas  = (const float*)d_in[7];
  const float* att0s  = (const float*)d_in[8];
  const float* Wout_s = (const float*)d_in[9];
  const float* bout_s = (const float*)d_in[10];
  const float* bnw1   = (const float*)d_in[11];
  const float* bnb1   = (const float*)d_in[12];
  const float* Wff_s  = (const float*)d_in[13];
  const float* bff_s  = (const float*)d_in[14];
  const float* bnw2   = (const float*)d_in[15];
  const float* bnb2   = (const float*)d_in[16];
  const float* Win_t  = (const float*)d_in[17];
  const float* bin_t  = (const float*)d_in[18];
  const float* alphat = (const float*)d_in[19];
  const float* x1_nett= (const float*)d_in[20];
  const float* betat  = (const float*)d_in[21];
  const float* att0t  = (const float*)d_in[22];
  const float* Wout_t = (const float*)d_in[23];
  const float* bout_t = (const float*)d_in[24];
  const float* bnw3   = (const float*)d_in[25];
  const float* bnb3   = (const float*)d_in[26];
  const float* Wff_t  = (const float*)d_in[27];
  const float* bff_t  = (const float*)d_in[28];
  const float* bnw4   = (const float*)d_in[29];
  const float* bnb4   = (const float*)d_in[30];

  char* ws = (char*)d_ws;
  bf16*  xb    = (bf16*)(ws + OFF_XB);     // x, later y2 (in place)
  bf16*  ybuf  = (bf16*)(ws + OFF_Y);
  bf16*  stage = (bf16*)(ws + OFF_STAGE);
  float* attb  = (float*)(ws + OFF_ATT);
  bf16*  x1sb  = (bf16*)(ws + OFF_X1S);
  bf16*  x1tb  = (bf16*)(ws + OFF_X1T);
  float* qkpes = (float*)(ws + OFF_QKPES);
  float* qkpet = (float*)(ws + OFF_QKPET);
  bf16*  WinSb = (bf16*)(ws + OFF_WINS);
  bf16*  WinTb = (bf16*)(ws + OFF_WINT);
  bf16*  WoSb  = (bf16*)(ws + OFF_WOUTS);
  bf16*  WoTb  = (bf16*)(ws + OFF_WOUTT);
  bf16*  WfSb  = (bf16*)(ws + OFF_WFFS);
  bf16*  WfTb  = (bf16*)(ws + OFF_WFFT);
  float* stats = (float*)(ws + OFF_STATS); // 4 stages x 128

  const int SM_ATT = 311296;   // 64x1600 + 2x(16x1600) bf16 + 32x64 bf16 (<= 320KB WGP LDS)
  const int SM_YS  = 140800;
  const int SM_YT  = 184832;
  const int SM_FF  = 59904;
  (void)hipFuncSetAttribute((const void*)k_att<false>, hipFuncAttributeMaxDynamicSharedMemorySize, SM_ATT);
  (void)hipFuncSetAttribute((const void*)k_att<true>,  hipFuncAttributeMaxDynamicSharedMemorySize, SM_ATT);
  (void)hipFuncSetAttribute((const void*)k_y_s, hipFuncAttributeMaxDynamicSharedMemorySize, SM_YS);
  (void)hipFuncSetAttribute((const void*)k_y_t, hipFuncAttributeMaxDynamicSharedMemorySize, SM_YT);
  (void)hipFuncSetAttribute((const void*)k_ff,  hipFuncAttributeMaxDynamicSharedMemorySize, SM_FF);

  // ---- prep ----
  k_zero<<<2, 256, 0, stream>>>(stats, 512);
  k_castf2b<<<4096, 256, 0, stream>>>(x, xb, (N_ * CTV_) / 4);
  k_castf2b<<<6, 256, 0, stream>>>(Win_s, WinSb, (96 * 64) / 4);
  k_castf2b<<<6, 256, 0, stream>>>(Win_t, WinTb, (96 * 64) / 4);
  k_castf2b<<<12, 256, 0, stream>>>(Wout_s, WoSb, (64 * 192) / 4);
  k_castf2b<<<12, 256, 0, stream>>>(Wout_t, WoTb, (64 * 192) / 4);
  k_castf2b<<<4, 256, 0, stream>>>(Wff_s, WfSb, (64 * 64) / 4);
  k_castf2b<<<4, 256, 0, stream>>>(Wff_t, WfTb, (64 * 64) / 4);
  k_castf2b<<<300, 256, 0, stream>>>(x1_nets, x1sb, (S_ * CTV_) / 4);
  k_castf2b<<<300, 256, 0, stream>>>(x1_nett, x1tb, (S_ * CTV_) / 4);
  k_qkpe<<<600, 256, 0, stream>>>(Win_s, pe_s, bin_s, qkpes);
  k_qkpe<<<600, 256, 0, stream>>>(Win_t, pe_t, bin_t, qkpet);

  // ---- spatial half ----
  k_att<false><<<dim3(S_, N_), 256, SM_ATT, stream>>>(xb, WinSb, qkpes, x1sb, att0s, alphas, betas, attb);
  k_y_s<<<dim3(8, N_), 256, SM_YS, stream>>>(xb, attb, WoSb, bout_s, stage, stats + 0);
  k_bn_act<false><<<4096, 256, 0, stream>>>(stage, xb, stats + 0, bnw1, bnb1, ybuf, nullptr);
  k_ff<<<dim3(4, N_), 256, SM_FF, stream>>>(ybuf, WfSb, bff_s, stage, stats + 128);
  k_bn_act<false><<<4096, 256, 0, stream>>>(stage, xb, stats + 128, bnw2, bnb2, xb, nullptr); // y2 in place

  // ---- temporal half ----
  k_att<true><<<dim3(S_, N_), 256, SM_ATT, stream>>>(xb, WinTb, qkpet, x1tb, att0t, alphat, betat, attb);
  k_y_t<<<dim3(4, N_), 256, SM_YT, stream>>>(xb, attb, WoTb, bout_t, stage, stats + 256);
  k_bn_act<false><<<4096, 256, 0, stream>>>(stage, xb, stats + 256, bnw3, bnb3, ybuf, nullptr); // z1
  k_ff<<<dim3(4, N_), 256, SM_FF, stream>>>(ybuf, WfTb, bff_t, stage, stats + 384);
  k_bn_act<true><<<4096, 256, 0, stream>>>(stage, xb, stats + 384, bnw4, bnb4, nullptr, (float*)d_out);

  (void)in_sizes; (void)n_in; (void)out_size; (void)ws_size;
}
